// StateSpaceModelAttentionWithSSD_45818711113782
// MI455X (gfx1250) — compile-verified, matrix operand order checked by reading, not run on hardware
//
#include <hip/hip_runtime.h>

// out = x @ M, where M = (Wq*diag(A) + Wk*diag(B) + Wv*diag(C)) @ Wo.
// Single 32768x448x448 GEMM; bf16 WMMA with hi/lo f32-error compensation.

typedef __attribute__((ext_vector_type(16))) __bf16 v16bf;
typedef __attribute__((ext_vector_type(8)))  __bf16 v8bf;
typedef __attribute__((ext_vector_type(8)))  float  v8f;
typedef __attribute__((ext_vector_type(4)))  float  v4f;

#define DDIM 448

// ---------------------------------------------------------------------------
// Phase 1: build Mt_hi / Mt_lo = bf16 hi/lo split of transpose of
//   M = (Wq*diag(a) + Wk*diag(b) + Wv*diag(c)) @ Wo      (448x448, ~trivial)
// Stored transposed (Mt[n][k]) so B-fragment loads in phase 2 are contiguous.
// ---------------------------------------------------------------------------
__global__ void fuse_weights_kernel(const float* __restrict__ Wq,
                                    const float* __restrict__ Wk,
                                    const float* __restrict__ Wv,
                                    const float* __restrict__ Wo,
                                    const float* __restrict__ A,
                                    const float* __restrict__ B,
                                    const float* __restrict__ C,
                                    __bf16* __restrict__ MtHi,
                                    __bf16* __restrict__ MtLo) {
  __shared__ float Ts[16][17];
  __shared__ float Ws[16][17];
  const int tx = threadIdx.x, ty = threadIdx.y;
  const int i0 = blockIdx.y * 16;   // row of M  (k index of GEMM)
  const int o0 = blockIdx.x * 16;   // col of M  (n index of GEMM)
  float acc = 0.f;
  for (int kk = 0; kk < DDIM; kk += 16) {
    const int j = kk + tx;
    const float da = A[j * DDIM + j];
    const float db = B[j * DDIM + j];
    const float dc = C[j * DDIM + j];
    const int r = (i0 + ty) * DDIM + j;
    Ts[ty][tx] = Wq[r] * da + Wk[r] * db + Wv[r] * dc;
    Ws[ty][tx] = Wo[(kk + ty) * DDIM + (o0 + tx)];
    __syncthreads();
#pragma unroll
    for (int t = 0; t < 16; ++t) acc += Ts[ty][t] * Ws[t][tx];
    __syncthreads();
  }
  const __bf16 hi = (__bf16)acc;
  const __bf16 lo = (__bf16)(acc - (float)hi);
  const int idx = (o0 + tx) * DDIM + (i0 + ty);   // transposed store
  MtHi[idx] = hi;
  MtLo[idx] = lo;
}

// ---------------------------------------------------------------------------
// Phase 2: out[32768 x 448] = x[32768 x 448] @ M[448 x 448]
// Block: 256 threads (8 waves). Wave w -> 16-row strip; block -> 128 rows x
// 112 cols (7 col-tiles of 16). Mt slab staged once into LDS (196 KB/WGP).
// Accumulate in f32 via 3 bf16 WMMAs: Ahi*Bhi + Ahi*Blo + Alo*Bhi.
// ---------------------------------------------------------------------------
__global__ void __launch_bounds__(256)
ssm_fused_gemm_kernel(const float* __restrict__ x,
                      const __bf16* __restrict__ MtHi,
                      const __bf16* __restrict__ MtLo,
                      float* __restrict__ out) {
  __shared__ __bf16 Bsh[2][112 * DDIM];   // [hi|lo] slabs: 2 * 100,352 B

  const int lane = threadIdx.x & 31;
  const int wave = threadIdx.x >> 5;
  const int half = lane >> 4;      // lane group 0: K 0-7/16-23; group 1: 8-15/24-31
  const int l16  = lane & 15;
  const int rowBase = blockIdx.x * 128 + wave * 16;
  const int nBase   = blockIdx.y * 112;

  // ---- stage the 112-column Mt slab (all K) into LDS, cooperatively
  {
    const uint4* srcH = (const uint4*)(MtHi + (size_t)nBase * DDIM);
    const uint4* srcL = (const uint4*)(MtLo + (size_t)nBase * DDIM);
    uint4* dstH = (uint4*)&Bsh[0][0];
    uint4* dstL = (uint4*)&Bsh[1][0];
    const int nVec = 112 * DDIM * 2 / 16;   // 6272 16-byte vectors per slab
    for (int i = threadIdx.x; i < nVec; i += 256) {
      dstH[i] = srcH[i];
      dstL[i] = srcL[i];
    }
  }
  __syncthreads();

  v8f acc[7] = {};

  const float* aRow = x + (size_t)(rowBase + l16) * DDIM;

  for (int k = 0; k < DDIM; k += 32) {
    const int ka = k + half * 8;

    // ---- A fragment: 16 f32 -> bf16 hi + residual lo (A 16x32 layout)
    float fa[16];
    *(v4f*)(&fa[0])  = *(const v4f*)(aRow + ka);
    *(v4f*)(&fa[4])  = *(const v4f*)(aRow + ka + 4);
    *(v4f*)(&fa[8])  = *(const v4f*)(aRow + ka + 16);
    *(v4f*)(&fa[12]) = *(const v4f*)(aRow + ka + 20);
    if (k + 32 < DDIM) __builtin_prefetch(aRow + ka + 32, 0, 1);

    v16bf ahi, alo;
#pragma unroll
    for (int e = 0; e < 16; ++e) {
      const __bf16 h = (__bf16)fa[e];
      ahi[e] = h;
      alo[e] = (__bf16)(fa[e] - (float)h);
    }

    // ---- 7 col-tiles: B fragments from LDS, 3 compensated WMMAs each
#pragma unroll
    for (int t = 0; t < 7; ++t) {
      const int bo = (t * 16 + l16) * DDIM + ka;
      const v8bf h0 = *(const v8bf*)(&Bsh[0][bo]);
      const v8bf h1 = *(const v8bf*)(&Bsh[0][bo + 16]);
      const v8bf l0 = *(const v8bf*)(&Bsh[1][bo]);
      const v8bf l1 = *(const v8bf*)(&Bsh[1][bo + 16]);
      v16bf bhi, blo;
#pragma unroll
      for (int e = 0; e < 8; ++e) {
        bhi[e] = h0[e]; bhi[8 + e] = h1[e];
        blo[e] = l0[e]; blo[8 + e] = l1[e];
      }
      acc[t] = __builtin_amdgcn_wmma_f32_16x16x32_bf16(
          false, alo, false, bhi, (short)0, acc[t], false, false);
      acc[t] = __builtin_amdgcn_wmma_f32_16x16x32_bf16(
          false, ahi, false, blo, (short)0, acc[t], false, false);
      acc[t] = __builtin_amdgcn_wmma_f32_16x16x32_bf16(
          false, ahi, false, bhi, (short)0, acc[t], false, false);
    }
  }

  // ---- epilogue: C/D layout -> VGPR r holds M=r (lanes 0-15) / M=r+8
#pragma unroll
  for (int t = 0; t < 7; ++t) {
    float* o = out + (size_t)(rowBase + half * 8) * DDIM + nBase + t * 16 + l16;
#pragma unroll
    for (int r = 0; r < 8; ++r) o[(size_t)r * DDIM] = acc[t][r];
  }
}

// ---------------------------------------------------------------------------
extern "C" void kernel_launch(void* const* d_in, const int* in_sizes, int n_in,
                              void* d_out, int out_size, void* d_ws, size_t ws_size,
                              hipStream_t stream) {
  const float* x  = (const float*)d_in[0];
  const float* Wq = (const float*)d_in[1];
  const float* Wk = (const float*)d_in[2];
  const float* Wv = (const float*)d_in[3];
  const float* Wo = (const float*)d_in[4];
  const float* A  = (const float*)d_in[5];
  const float* B  = (const float*)d_in[6];
  const float* C  = (const float*)d_in[7];

  __bf16* MtHi = (__bf16*)d_ws;                 // 448*448*2 = 401,408 B
  __bf16* MtLo = MtHi + (size_t)DDIM * DDIM;    // another 401,408 B

  fuse_weights_kernel<<<dim3(28, 28), dim3(16, 16), 0, stream>>>(
      Wq, Wk, Wv, Wo, A, B, C, MtHi, MtLo);

  const int rows  = in_sizes[0] / DDIM;         // 8*4096 = 32768
  const int gridX = rows / 128;                 // 256
  ssm_fused_gemm_kernel<<<dim3(gridX, 4), 256, 0, stream>>>(
      x, MtHi, MtLo, (float*)d_out);
}
